// AttnLRPAttention_24060406792825
// MI455X (gfx1250) — matrix operand order repaired and checked by reading, not compiled, not run on hardware
//
#include <hip/hip_runtime.h>

typedef _Float16 f16;
typedef _Float16 v16h __attribute__((ext_vector_type(16)));
typedef _Float16 v8h  __attribute__((ext_vector_type(8)));
typedef float    v8f  __attribute__((ext_vector_type(8)));

#define WMMA_F32_F16(A, B, C) \
  __builtin_amdgcn_wmma_f32_16x16x32_f16(false, (A), false, (B), (short)0, (C), false, false)

static __device__ __forceinline__ v16h cat16(v8h lo, v8h hi) {
  return __builtin_shufflevector(lo, hi, 0,1,2,3,4,5,6,7,8,9,10,11,12,13,14,15);
}

// ---- WMMA operand loaders (gfx1250 wave32 layouts, ISA 7.12.2) ----
static __device__ __forceinline__ v16h load_a_f16(const f16* base, int ld, int lane) {
  const f16* p = base + (lane & 15) * ld + ((lane < 16) ? 0 : 8);
  return cat16(*(const v8h*)p, *(const v8h*)(p + 16));
}
static __device__ __forceinline__ v16h load_bt_f16(const f16* base, int ld, int lane) {
  const f16* p = base + (lane & 15) * ld + ((lane < 16) ? 0 : 8);
  return cat16(*(const v8h*)p, *(const v8h*)(p + 16));  // 2x16B, 16B-aligned at all call sites
}

// ---- CDNA5 async global->LDS copy (ASYNCcnt path, no VGPR round-trip) ----
// Flat pointers into LDS carry the LDS byte offset in addr[31:0] (ISA 10.2 aperture rules).
static __device__ __forceinline__ void async_b128(const f16* g, f16* l) {
  unsigned lds_addr = (unsigned)(unsigned long long)(void*)l;
  asm volatile("global_load_async_to_lds_b128 %0, %1, off"
               :: "v"(lds_addr), "v"(g)
               : "memory");
}
static __device__ __forceinline__ void wait_async0() {
#if __has_builtin(__builtin_amdgcn_s_wait_asynccnt)
  __builtin_amdgcn_s_wait_asynccnt(0);
#else
  asm volatile("s_wait_asynccnt 0x0" ::: "memory");
#endif
}

// ---------------- Kernel 0: f32 -> f16 bulk convert ----------------
__global__ __launch_bounds__(256) void cvt_kernel(const float* __restrict__ s,
                                                  f16* __restrict__ d, int n8) {
  const int i = blockIdx.x * 256 + threadIdx.x;
  if (i < n8) {
    v8f x = *(const v8f*)(s + (size_t)i * 8);
    *(v8h*)(d + (size_t)i * 8) = __builtin_convertvector(x, v8h);
  }
}

// ---------------- Kernel 1: QKV projection ----------------
// out[m][c] = sum_k xh[m][k] * w[c][k] + bias[c]; per-wave 32x64 tile.
// B panel double-buffered in LDS via async loads.
__global__ __launch_bounds__(256) void qkv_kernel(
    const f16* __restrict__ xh, const f16* __restrict__ w, const float* __restrict__ bias,
    f16* __restrict__ q, f16* __restrict__ kk, f16* __restrict__ vt) {
  __shared__ f16 lb[2][64 * 32];                 // B panel: 64 out-cols x 32 k, x2 buffers
  const int tid = threadIdx.x;
  const int lane = tid & 31, wave = tid >> 5;
  const int row0 = (blockIdx.x * 8 + wave) * 32;
  const int col0 = blockIdx.y * 64;

  // per-thread staging slot: 16B (row pr, cols pc..pc+7 of the 64x32 panel)
  const int pr = tid >> 2, pc = (tid & 3) * 8;
  const f16* wsrc = w + (size_t)(col0 + pr) * 1024 + pc;
  f16* ld0 = &lb[0][pr * 32 + pc];
  f16* ld1 = &lb[1][pr * 32 + pc];

  async_b128(wsrc, ld0);                         // prologue: k0 = 0 -> buf 0
  v8f acc[2][4] = {};
  for (int k0 = 0; k0 < 1024; k0 += 32) {
    const int cur = (k0 >> 5) & 1;
    wait_async0();                               // my writes to buf[cur] done
    __syncthreads();                             // everyone's writes done; prev reads done
    if (k0 + 32 < 1024)
      async_b128(wsrc + k0 + 32, cur ? ld0 : ld1);
    v16h a0 = load_a_f16(xh + (size_t)row0 * 1024 + k0, 1024, lane);
    v16h a1 = load_a_f16(xh + (size_t)(row0 + 16) * 1024 + k0, 1024, lane);
#pragma unroll
    for (int t = 0; t < 4; ++t) {
      v16h b = load_bt_f16(&lb[cur][(16 * t) * 32], 32, lane);
      acc[0][t] = WMMA_F32_F16(a0, b, acc[0][t]);
      acc[1][t] = WMMA_F32_F16(a1, b, acc[1][t]);
    }
  }
  const int rbase = (lane < 16) ? 0 : 8;
  const int bb = row0 >> 11;
  const int nbase = row0 & 2047;
#pragma unroll
  for (int t = 0; t < 4; ++t) {
    const int c = col0 + 16 * t + (lane & 15);
    const float bs = bias[c];
    const int s = c >> 10;                       // 0=q 1=k 2=v (uniform per wave)
    const int h = (c & 1023) >> 6;
    const int d = c & 63;
    const size_t bh = (size_t)(bb * 16 + h);
#pragma unroll
    for (int mh = 0; mh < 2; ++mh) {
      if (s == 2) {                              // v: packed 16B store of 8 consecutive tokens
        v8h pk;
#pragma unroll
        for (int r = 0; r < 8; ++r) pk[r] = (f16)(acc[mh][t][r] + bs);
        *(v8h*)(vt + (bh * 64 + d) * 2048 + nbase + 16 * mh + rbase) = pk;
      } else {
#pragma unroll
        for (int r = 0; r < 8; ++r) {
          const int n = nbase + 16 * mh + rbase + r;
          const float v = acc[mh][t][r] + bs;
          if (s == 0) q [(bh * 2048 + n) * 64 + d] = (f16)(v * 0.125f);
          else        kk[(bh * 2048 + n) * 64 + d] = (f16)v;
        }
      }
    }
  }
}

// ---------------- Kernel 2: flash attention ----------------
// Per wave: 16 queries; 64-key K/V blocks double-buffered in LDS via async loads.
// Computes S^T = K·Q^T so the score C-layout == P's A-operand layout.
__global__ __launch_bounds__(256) void attn_kernel(
    const f16* __restrict__ q, const f16* __restrict__ k,
    const f16* __restrict__ vt, f16* __restrict__ o) {
  __shared__ f16 lk[2][64 * 64];                 // [key][d]
  __shared__ f16 lv[2][64 * 64];                 // [d][key]
  const int tid = threadIdx.x;
  const int lane = tid & 31, wave = tid >> 5;
  const int bh = blockIdx.y;                     // b*16 + h
  const int q0 = blockIdx.x * 128 + wave * 16;
  const f16* qp = q  + (size_t)bh * (2048 * 64);
  const f16* kp = k  + (size_t)bh * (2048 * 64);
  const f16* vp = vt + (size_t)bh * (64 * 2048);

  // staging slots (per thread: 2x16B of K, 2x16B of V)
  const int vr = tid >> 2, vc = (tid & 3) * 16;
  const f16* ksrc = kp + (size_t)tid * 8;        // + kb0*64 per call
  const f16* vsrc = vp + (size_t)vr * 2048 + vc; // + kb0 per call

  // Q^T as B operand, loaded once
  const v16h qb0 = load_bt_f16(qp + (size_t)q0 * 64 + 0,  64, lane);
  const v16h qb1 = load_bt_f16(qp + (size_t)q0 * 64 + 32, 64, lane);

  // prologue: stage block kb0 = 0 into buf 0
  async_b128(ksrc,        &lk[0][tid * 8]);
  async_b128(ksrc + 2048, &lk[0][tid * 8 + 2048]);
  async_b128(vsrc,        &lv[0][vr * 64 + vc]);
  async_b128(vsrc + 8,    &lv[0][vr * 64 + vc + 8]);

  v8f oacc[4] = {};
  float m_run = -1e30f, l_run = 0.f;             // lane owns stats for query q0+(lane&15)
  const int rbase = (lane < 16) ? 0 : 8;

  for (int kb0 = 0; kb0 < 2048; kb0 += 64) {
    const int cur = (kb0 >> 6) & 1;
    wait_async0();
    __syncthreads();
    if (kb0 + 64 < 2048) {                       // stage next block into other buffer
      const int nxt = cur ^ 1;
      const f16* ks = ksrc + (size_t)(kb0 + 64) * 64;
      const f16* vs = vsrc + (kb0 + 64);
      async_b128(ks,        &lk[nxt][tid * 8]);
      async_b128(ks + 2048, &lk[nxt][tid * 8 + 2048]);
      async_b128(vs,        &lv[nxt][vr * 64 + vc]);
      async_b128(vs + 8,    &lv[nxt][vr * 64 + vc + 8]);
    }

    // S^T: 4 tiles of (16 keys x 16 queries), reduced over 64 head dims
    v8f s[4];
#pragma unroll
    for (int t = 0; t < 4; ++t) {
      const f16* kr = &lk[cur][(16 * t) * 64];
      v16h a0 = load_a_f16(kr + 0,  64, lane);
      v16h a1 = load_a_f16(kr + 32, 64, lane);
      v8f c = {};
      c = WMMA_F32_F16(a0, qb0, c);
      c = WMMA_F32_F16(a1, qb1, c);
      s[t] = c;
    }
    // online softmax
    float mx = -1e30f;
#pragma unroll
    for (int t = 0; t < 4; ++t)
#pragma unroll
      for (int r = 0; r < 8; ++r) mx = fmaxf(mx, s[t][r]);
    mx = fmaxf(mx, __shfl_xor(mx, 16, 32));
    const float m_new = fmaxf(m_run, mx);
    if (__any(mx > m_run)) {                     // wave-uniform; corr==1 exactly when skipped
      const float corr = __expf(m_run - m_new);
      l_run *= corr;
#pragma unroll
      for (int r = 0; r < 8; ++r) {
        const float f = __shfl(corr, rbase + r, 32);
        oacc[0][r] *= f; oacc[1][r] *= f; oacc[2][r] *= f; oacc[3][r] *= f;
      }
    }
    m_run = m_new;
    float rs = 0.f;
#pragma unroll
    for (int t = 0; t < 4; ++t)
#pragma unroll
      for (int r = 0; r < 8; ++r) {
        const float e = __expf(s[t][r] - m_new);
        s[t][r] = e;
        rs += e;
      }
    rs += __shfl_xor(rs, 16, 32);
    l_run += rs;

    // pack P: two consecutive S^T C-tiles == one A operand (K = 32 keys)
    v16h pa0, pa1;
#pragma unroll
    for (int r = 0; r < 8; ++r) {
      pa0[r]     = (f16)s[0][r];
      pa0[8 + r] = (f16)s[1][r];
      pa1[r]     = (f16)s[2][r];
      pa1[8 + r] = (f16)s[3][r];
    }
    // O += P·V from LDS
#pragma unroll
    for (int dt = 0; dt < 4; ++dt) {
      v16h b0 = load_bt_f16(&lv[cur][(16 * dt) * 64 + 0],  64, lane);
      v16h b1 = load_bt_f16(&lv[cur][(16 * dt) * 64 + 32], 64, lane);
      oacc[dt] = WMMA_F32_F16(pa0, b0, oacc[dt]);
      oacc[dt] = WMMA_F32_F16(pa1, b1, oacc[dt]);
    }
  }
  // normalize, store to [B][N][H*D] f16
  const float inv = 1.f / l_run;
  const int b = bh >> 4, h = bh & 15;
#pragma unroll
  for (int r = 0; r < 8; ++r) {
    const float f = __shfl(inv, rbase + r, 32);
    const int row = q0 + rbase + r;
#pragma unroll
    for (int dt = 0; dt < 4; ++dt) {
      const int cc = h * 64 + dt * 16 + (lane & 15);
      o[((size_t)b * 2048 + row) * 1024 + cc] = (f16)(oacc[dt][r] * f);
    }
  }
}

// ---------------- Kernel 3: output projection ----------------
__global__ __launch_bounds__(256) void proj_kernel(
    const f16* __restrict__ a, const f16* __restrict__ w,
    const float* __restrict__ bias, float* __restrict__ y) {
  __shared__ f16 lb[2][64 * 32];
  const int tid = threadIdx.x;
  const int lane = tid & 31, wave = tid >> 5;
  const int row0 = (blockIdx.x * 8 + wave) * 32;
  const int col0 = blockIdx.y * 64;

  const int pr = tid >> 2, pc = (tid & 3) * 8;
  const f16* wsrc = w + (size_t)(col0 + pr) * 1024 + pc;
  f16* ld0 = &lb[0][pr * 32 + pc];
  f16* ld1 = &lb[1][pr * 32 + pc];

  async_b128(wsrc, ld0);
  v8f acc[2][4] = {};
  for (int k0 = 0; k0 < 1024; k0 += 32) {
    const int cur = (k0 >> 5) & 1;
    wait_async0();
    __syncthreads();
    if (k0 + 32 < 1024)
      async_b128(wsrc + k0 + 32, cur ? ld0 : ld1);
    v16h a0 = load_a_f16(a + (size_t)row0 * 1024 + k0, 1024, lane);
    v16h a1 = load_a_f16(a + (size_t)(row0 + 16) * 1024 + k0, 1024, lane);
#pragma unroll
    for (int t = 0; t < 4; ++t) {
      v16h b = load_bt_f16(&lb[cur][(16 * t) * 32], 32, lane);
      acc[0][t] = WMMA_F32_F16(a0, b, acc[0][t]);
      acc[1][t] = WMMA_F32_F16(a1, b, acc[1][t]);
    }
  }
  const int rbase = (lane < 16) ? 0 : 8;
#pragma unroll
  for (int t = 0; t < 4; ++t) {
    const int c = col0 + 16 * t + (lane & 15);
    const float bs = bias[c];
#pragma unroll
    for (int mh = 0; mh < 2; ++mh)
#pragma unroll
      for (int r = 0; r < 8; ++r)
        y[(size_t)(row0 + 16 * mh + rbase + r) * 1024 + c] = acc[mh][t][r] + bs;
  }
}

extern "C" void kernel_launch(void* const* d_in, const int* in_sizes, int n_in,
                              void* d_out, int out_size, void* d_ws, size_t ws_size,
                              hipStream_t stream) {
  (void)in_sizes; (void)n_in; (void)out_size; (void)ws_size;
  const float* x      = (const float*)d_in[0];
  const float* w_qkv  = (const float*)d_in[1];
  const float* b_qkv  = (const float*)d_in[2];
  const float* w_proj = (const float*)d_in[3];
  const float* b_proj = (const float*)d_in[4];
  float* out = (float*)d_out;

  const size_t QKV = (size_t)8 * 1024 * 1024;    // B*H*N*D = 4*16*2048*64
  f16* xh  = (f16*)d_ws;                         //  8M
  f16* wqh = xh  + QKV;                          //  3M
  f16* wph = wqh + (size_t)3 * 1024 * 1024;      //  1M
  f16* q   = wph + (size_t)1 * 1024 * 1024;      //  8M
  f16* kk  = q   + QKV;                          //  8M
  f16* vt  = kk  + QKV;                          //  8M
  f16* ao  = vt  + QKV;                          //  8M  (total 88 MB)

  cvt_kernel<<<dim3(4096), 256, 0, stream>>>(x,      xh,  1048576);
  cvt_kernel<<<dim3(1536), 256, 0, stream>>>(w_qkv,  wqh, 393216);
  cvt_kernel<<<dim3(512),  256, 0, stream>>>(w_proj, wph, 131072);
  qkv_kernel <<<dim3(32, 48), 256, 0, stream>>>(xh, wqh, b_qkv, q, kk, vt);
  attn_kernel<<<dim3(16, 64), 256, 0, stream>>>(q, kk, vt, ao);
  proj_kernel<<<dim3(32, 16), 256, 0, stream>>>(ao, wph, b_proj, out);
}
